// EquiformerWEdgesBackbone_31671088840832
// MI455X (gfx1250) — matrix operand order repaired and checked by reading, not compiled, not run on hardware
//
#include <hip/hip_runtime.h>
#include <hip/hip_bf16.h>

#define NN 6000
#define NE 150000
#define KC 9
#define CC 128
#define HH 8
#define LL 2
#define NB 512
#define NKROWS (NN * KC)                  /* 54000  */
#define RROWS  (NE * KC)                  /* 1350000 */

typedef _Float16 half_t;
typedef half_t h8v  __attribute__((ext_vector_type(8)));
typedef half_t h16v __attribute__((ext_vector_type(16)));
typedef float  f8v  __attribute__((ext_vector_type(8)));

union H16U { h16v v; h8v h[2]; half_t e[16]; };

__device__ __forceinline__ h16v ld2x8(const half_t* p0, const half_t* p1) {
  H16U u;
  u.h[0] = *reinterpret_cast<const h8v*>(p0);
  u.h[1] = *reinterpret_cast<const h8v*>(p1);
  return u.v;
}

__device__ __forceinline__ f8v wmma16(h16v a, h16v b, f8v c) {
  return __builtin_amdgcn_wmma_f32_16x16x32_f16(false, a, false, b, (short)0, c,
                                                false, false);
}

__device__ __forceinline__ float sigm(float x) { return 1.f / (1.f + __expf(-x)); }
__device__ __forceinline__ float silu(float x) { return x * sigm(x); }

__device__ __forceinline__ unsigned f2ord(float f) {
  unsigned u = __float_as_uint(f);
  return (u & 0x80000000u) ? ~u : (u | 0x80000000u);
}
__device__ __forceinline__ float ord2f(unsigned o) {
  unsigned u = (o & 0x80000000u) ? (o & 0x7fffffffu) : ~o;
  return __uint_as_float(u);
}

/* ------------------------- elementwise / utility ------------------------- */

__global__ void k_w_cm(const float* __restrict__ w, half_t* __restrict__ o,
                       int Kd, int Nd) {
  int idx = blockIdx.x * blockDim.x + threadIdx.x;
  if (idx >= Kd * Nd) return;
  int k = idx / Nd, n = idx % Nd;
  o[(long long)n * Kd + k] = (half_t)w[idx];
}

__global__ void k_zero_f32(float* __restrict__ p, int n) {
  int i = blockIdx.x * blockDim.x + threadIdx.x;
  if (i < n) p[i] = 0.f;
}

__global__ void k_f32_to_f16(const float* __restrict__ in, half_t* __restrict__ o, int n) {
  int i = blockIdx.x * blockDim.x + threadIdx.x;
  if (i < n) o[i] = (half_t)in[i];
}

__global__ void k_node_init(const int* __restrict__ atom_feats,
                            const float* __restrict__ atom_emb,
                            float* __restrict__ x) {
  int idx = blockIdx.x * blockDim.x + threadIdx.x;     // N*128
  if (idx >= NN * CC) return;
  int n = idx >> 7, c = idx & 127;
  float a = 0.f;
  #pragma unroll
  for (int f = 0; f < 4; ++f) {
    int id = atom_feats[n * 4 + f];
    a += atom_emb[(f * 10 + id) * CC + c];
  }
  long long base = (long long)n * KC * CC + c;
  x[base] = a;
  #pragma unroll
  for (int k = 1; k < KC; ++k) x[base + k * CC] = 0.f;
}

/* rms_norm_sh over (K,C) per node -> f16 */
__global__ void k_rms_f16(const float* __restrict__ x, half_t* __restrict__ o) {
  __shared__ float red[256];
  int n = blockIdx.x, t = threadIdx.x;
  const float* xp = x + (long long)n * (KC * CC);
  float s = 0.f;
  for (int i = t; i < KC * CC; i += 256) { float v = xp[i]; s += v * v; }
  red[t] = s; __syncthreads();
  for (int w = 128; w > 0; w >>= 1) { if (t < w) red[t] += red[t + w]; __syncthreads(); }
  float sc = rsqrtf(red[0] / (float)(KC * CC) + 1e-6f);
  half_t* op = o + (long long)n * (KC * CC);
  for (int i = t; i < KC * CC; i += 256) op[i] = (half_t)(xp[i] * sc);
}

__global__ void k_rms_out(const float* __restrict__ x, float* __restrict__ o) {
  __shared__ float red[256];
  int n = blockIdx.x, t = threadIdx.x;
  const float* xp = x + (long long)n * (KC * CC);
  float s = 0.f;
  for (int i = t; i < KC * CC; i += 256) { float v = xp[i]; s += v * v; }
  red[t] = s; __syncthreads();
  for (int w = 128; w > 0; w >>= 1) { if (t < w) red[t] += red[t + w]; __syncthreads(); }
  float sc = rsqrtf(red[0] / (float)(KC * CC) + 1e-6f);
  float* op = o + (long long)n * (KC * CC);
  for (int i = t; i < KC * CC; i += 256) op[i] = xp[i] * sc;
}

/* ------------------------------ WMMA GEMMs ------------------------------ */
/* All GEMM waves compute a 16-row x 128-col output block with 8 f32x8
 * accumulators: A is loaded once per K-step and reused across 8 WMMAs,
 * so the WMMA:VMEM ratio is 8 wmma : 18 b128-loads per K-step. */

/* generic: out[r, n0+0..127] (=|+=) A[r,:KD] @ Bcm; grid=(rows/16, Nd/128) */
template <int KD>
__global__ __launch_bounds__(32)
void k_gemm_plain8(const half_t* __restrict__ A,
                   const half_t* __restrict__ Bcm,
                   float* __restrict__ out, int Nd, int accumulate) {
  const int lane = threadIdx.x, hi = lane >> 4, lm = lane & 15;
  const long long row0 = (long long)blockIdx.x * 16;
  const int n0 = blockIdx.y * 128;
  const half_t* arow  = A + (row0 + lm) * (long long)KD;
  const half_t* bbase = Bcm + (long long)(n0 + lm) * KD + hi * 8;
  __builtin_prefetch(arow, 0, 0);
  f8v acc[8] = {};
  for (int kb = 0; kb < KD / 32; ++kb) {
    const half_t* ap = arow + kb * 32 + hi * 8;
    h16v a = ld2x8(ap, ap + 16);
    #pragma unroll
    for (int t = 0; t < 8; ++t) {
      const half_t* bp = bbase + (long long)t * 16 * KD + kb * 32;
      h16v b = ld2x8(bp, bp + 16);
      acc[t] = wmma16(a, b, acc[t]);
    }
  }
  #pragma unroll
  for (int t = 0; t < 8; ++t) {
    float* op = out + row0 * Nd + n0 + t * 16 + lm;
    #pragma unroll
    for (int r = 0; r < 8; ++r) {
      int m = r + 8 * hi;
      if (accumulate) op[(long long)m * Nd] += acc[t][r];
      else            op[(long long)m * Nd]  = acc[t][r];
    }
  }
}

/* edge MLP stage 1: silu(rbf(dist) @ W1 + b1) -> f16 [E,128]; rbf in-register */
__global__ __launch_bounds__(32)
void k_gemm_rbf8(const float* __restrict__ dist,
                 const half_t* __restrict__ Bcm,    /* [128n x 512k] */
                 const float* __restrict__ bias,
                 half_t* __restrict__ outH) {
  const int lane = threadIdx.x, hi = lane >> 4, lm = lane & 15;
  const long long row0 = (long long)blockIdx.x * 16;
  const float d = dist[row0 + lm];
  const float cstep = 5.0f / 511.0f;     /* linspace(0,RMAX,512) */
  const float invw  = 512.0f / 5.0f;     /* 1/width, width=RMAX/B */
  const half_t* bbase = Bcm + (long long)lm * NB + hi * 8;
  f8v acc[8] = {};
  for (int kb = 0; kb < NB / 32; ++kb) {
    H16U a;
    #pragma unroll
    for (int i = 0; i < 16; ++i) {
      int k = kb * 32 + hi * 8 + (i < 8 ? i : i + 8);
      float t = (d - (float)k * cstep) * invw;
      a.e[i] = (half_t)__expf(-t * t);
    }
    #pragma unroll
    for (int t = 0; t < 8; ++t) {
      const half_t* bp = bbase + (long long)t * 16 * NB + kb * 32;
      h16v b = ld2x8(bp, bp + 16);
      acc[t] = wmma16(a.v, b, acc[t]);
    }
  }
  #pragma unroll
  for (int t = 0; t < 8; ++t) {
    int n = t * 16 + lm;
    float bv = bias[n];
    #pragma unroll
    for (int r = 0; r < 8; ++r) {
      int m = r + 8 * hi;
      outH[(row0 + m) * CC + n] = (half_t)silu(acc[t][r] + bv);
    }
  }
}

/* attention stage 1: silu([xn[src,0] | xn[dst,0] | e] @ Wa1) -> f16 [E,128] */
__global__ __launch_bounds__(32)
void k_gemm_attn18(const half_t* __restrict__ xn,   /* [N,9,128] */
                   const half_t* __restrict__ ef,   /* [E,128]   */
                   const int* __restrict__ src, const int* __restrict__ dst,
                   const half_t* __restrict__ Bcm,  /* [128n x 384k] */
                   half_t* __restrict__ outH) {
  const int lane = threadIdx.x, hi = lane >> 4, lm = lane & 15;
  const long long row0 = (long long)blockIdx.x * 16;
  const long long e = row0 + lm;
  const half_t* seg0 = xn + (long long)src[e] * (KC * CC);
  const half_t* seg1 = xn + (long long)dst[e] * (KC * CC);
  const half_t* seg2 = ef + e * CC;
  __builtin_prefetch(seg0, 0, 0);
  __builtin_prefetch(seg1, 0, 0);
  const half_t* bbase = Bcm + (long long)lm * (3 * CC) + hi * 8;
  f8v acc[8] = {};
  for (int kb = 0; kb < (3 * CC) / 32; ++kb) {
    int kg = kb * 32;
    const half_t* base = (kg < 128) ? seg0 : (kg < 256 ? seg1 : seg2);
    const half_t* ap = base + (kg & 127) + hi * 8;
    h16v a = ld2x8(ap, ap + 16);
    #pragma unroll
    for (int t = 0; t < 8; ++t) {
      const half_t* bp = bbase + (long long)t * 16 * (3 * CC) + kb * 32;
      h16v b = ld2x8(bp, bp + 16);
      acc[t] = wmma16(a, b, acc[t]);
    }
  }
  #pragma unroll
  for (int t = 0; t < 8; ++t) {
    int n = t * 16 + lm;
    #pragma unroll
    for (int r = 0; r < 8; ++r) {
      int m = r + 8 * hi;
      outH[(row0 + m) * CC + n] = (half_t)silu(acc[t][r]);
    }
  }
}

/* value GEMM over flattened (edge,k) rows, fused alpha*gate scale + scatter */
__global__ __launch_bounds__(32)
void k_gemm_value8(const half_t* __restrict__ xn,
                   const int* __restrict__ src, const int* __restrict__ dstA,
                   const float* __restrict__ scale,   /* [E,8] */
                   const half_t* __restrict__ Bcm,    /* Wv cm [128x128] */
                   float* __restrict__ agg) {         /* [N*9,128] */
  const int lane = threadIdx.x, hi = lane >> 4, lm = lane & 15;
  const long long row0 = (long long)blockIdx.x * 16;
  /* A row for this lane */
  long long r = row0 + lm;
  int e = (int)(r / 9), k = (int)(r - 9LL * e);
  const half_t* arow = xn + (long long)src[e] * (KC * CC) + k * CC;
  __builtin_prefetch(arow, 0, 0);
  /* epilogue row metadata for this lane's 8 output rows */
  int eo[8], ko[8], dn[8];
  #pragma unroll
  for (int i = 0; i < 8; ++i) {
    long long rr = row0 + i + 8 * hi;
    eo[i] = (int)(rr / 9);
    ko[i] = (int)(rr - 9LL * eo[i]);
    dn[i] = dstA[eo[i]];
  }
  const half_t* bbase = Bcm + (long long)lm * CC + hi * 8;
  f8v acc[8] = {};
  for (int kb = 0; kb < CC / 32; ++kb) {
    const half_t* ap = arow + kb * 32 + hi * 8;
    h16v a = ld2x8(ap, ap + 16);
    #pragma unroll
    for (int t = 0; t < 8; ++t) {
      const half_t* bp = bbase + (long long)t * 16 * CC + kb * 32;
      h16v b = ld2x8(bp, bp + 16);
      acc[t] = wmma16(a, b, acc[t]);
    }
  }
  #pragma unroll
  for (int t = 0; t < 8; ++t) {          /* head == t (V = 16) */
    int n = t * 16 + lm;
    #pragma unroll
    for (int i = 0; i < 8; ++i) {
      float sc = scale[(long long)eo[i] * HH + t];
      atomicAdd(&agg[((long long)dn[i] * KC + ko[i]) * CC + n], acc[t][i] * sc);
    }
  }
}

/* ------------------------- edge / softmax pieces ------------------------- */

__global__ void k_edge_final(const float* __restrict__ raw,
                             const float* __restrict__ b2,
                             const float* __restrict__ bond_emb,
                             const int* __restrict__ bond_feats,
                             half_t* __restrict__ ef) {
  int idx = blockIdx.x * blockDim.x + threadIdx.x;   /* E*128 */
  if (idx >= NE * CC) return;
  int e = idx >> 7, c = idx & 127;
  float v = raw[idx] + b2[c];
  #pragma unroll
  for (int f = 0; f < 3; ++f) {
    int id = bond_feats[e * 3 + f];
    v += bond_emb[(f * 8 + id) * CC + c];
  }
  ef[idx] = (half_t)v;
}

__global__ void k_seg_init(unsigned* __restrict__ segord, float* __restrict__ segsum) {
  int i = blockIdx.x * blockDim.x + threadIdx.x;
  if (i < NN * HH) { segord[i] = 0u; segsum[i] = 0.f; }
}

__global__ void k_logits_segmax(const half_t* __restrict__ hA,
                                const float* __restrict__ Wa2,  /* [128,8] layer slice */
                                const int* __restrict__ dst,
                                float* __restrict__ logits,
                                unsigned* __restrict__ segord) {
  int idx = blockIdx.x * blockDim.x + threadIdx.x;   /* E*8 */
  if (idx >= NE * HH) return;
  int e = idx >> 3, h = idx & 7;
  const half_t* row = hA + (long long)e * CC;
  float s = 0.f;
  #pragma unroll 8
  for (int c = 0; c < CC; ++c) s += (float)row[c] * Wa2[c * HH + h];
  logits[idx] = s;
  atomicMax(&segord[dst[e] * HH + h], f2ord(s));
}

__global__ void k_seg_expsum(const float* __restrict__ logits,
                             const unsigned* __restrict__ segord,
                             const int* __restrict__ dst,
                             float* __restrict__ scale,
                             float* __restrict__ segsum) {
  int idx = blockIdx.x * blockDim.x + threadIdx.x;
  if (idx >= NE * HH) return;
  int e = idx >> 3, h = idx & 7;
  float m = ord2f(segord[dst[e] * HH + h]);
  float ex = __expf(logits[idx] - m);
  scale[idx] = ex;
  atomicAdd(&segsum[dst[e] * HH + h], ex);
}

__global__ void k_scale_final(const half_t* __restrict__ ef,
                              const float* __restrict__ Wg,     /* [128,8] layer slice */
                              const int* __restrict__ dst,
                              const float* __restrict__ segsum,
                              float* __restrict__ scale) {
  int idx = blockIdx.x * blockDim.x + threadIdx.x;
  if (idx >= NE * HH) return;
  int e = idx >> 3, h = idx & 7;
  const half_t* row = ef + (long long)e * CC;
  float g = 0.f;
  #pragma unroll 8
  for (int c = 0; c < CC; ++c) g += (float)row[c] * Wg[c * HH + h];
  scale[idx] = scale[idx] / (segsum[dst[e] * HH + h] + 1e-9f) * sigm(g);
}

/* separable S2 activation: k=0 -> silu, k>0 gated by sigmoid(h[k=0]) */
__global__ void k_ffn_act(const float* __restrict__ hraw,  /* [N*9,512] */
                          half_t* __restrict__ hact) {
  int idx = blockIdx.x * blockDim.x + threadIdx.x;   /* N*512 */
  if (idx >= NN * 512) return;
  int n = idx >> 9, f = idx & 511;
  long long base = (long long)n * KC * 512 + f;
  float h0 = hraw[base];
  float sg = sigm(h0);
  hact[base] = (half_t)(h0 * sg);
  #pragma unroll
  for (int k = 1; k < KC; ++k)
    hact[base + (long long)k * 512] = (half_t)(hraw[base + (long long)k * 512] * sg);
}

/* -------------------------------- launch -------------------------------- */

extern "C" void kernel_launch(void* const* d_in, const int* in_sizes, int n_in,
                              void* d_out, int out_size, void* d_ws, size_t ws_size,
                              hipStream_t stream) {
  const int*   atom_feats = (const int*)d_in[0];
  const int*   bond_feats = (const int*)d_in[1];
  const int*   edge_index = (const int*)d_in[2];
  const float* edge_dist  = (const float*)d_in[3];
  const float* atom_emb   = (const float*)d_in[4];
  const float* bond_emb   = (const float*)d_in[5];
  const float* W_rbf1     = (const float*)d_in[6];
  const float* b_rbf1     = (const float*)d_in[7];
  const float* W_rbf2     = (const float*)d_in[8];
  const float* b_rbf2     = (const float*)d_in[9];
  const float* Wa1        = (const float*)d_in[10];
  const float* Wa2        = (const float*)d_in[11];
  const float* Wv         = (const float*)d_in[12];
  const float* Wg         = (const float*)d_in[13];
  const float* Wo         = (const float*)d_in[14];
  const float* Wf1        = (const float*)d_in[15];
  const float* Wf2        = (const float*)d_in[16];
  const int* src = edge_index;
  const int* dst = edge_index + NE;

  /* bump allocator on workspace */
  char* wsb = (char*)d_ws;
  size_t off = 0;
  auto alloc = [&](size_t bytes) -> char* {
    char* p = wsb + off;
    off = (off + bytes + 255) & ~(size_t)255;
    return p;
  };

  half_t* xn_f16  = (half_t*)alloc((size_t)NKROWS * CC * 2);
  half_t* e_f16   = (half_t*)alloc((size_t)NE * CC * 2);
  half_t* edge_h  = (half_t*)alloc((size_t)NE * CC * 2);
  float*  agg     = (float*)alloc((size_t)NKROWS * CC * 4);
  half_t* agg_f16 = (half_t*)alloc((size_t)NKROWS * CC * 2);
  float*  xbuf    = (float*)alloc((size_t)NKROWS * CC * 4);
  float*  tmp     = (float*)alloc((size_t)NKROWS * 512 * 4);   /* also edge raw */
  half_t* h_f16   = (half_t*)alloc((size_t)NKROWS * 512 * 2);
  float*  logits  = (float*)alloc((size_t)NE * HH * 4);
  float*  scale   = (float*)alloc((size_t)NE * HH * 4);
  unsigned* segord = (unsigned*)alloc((size_t)NN * HH * 4);
  float*  segsum  = (float*)alloc((size_t)NN * HH * 4);
  half_t* W1cm    = (half_t*)alloc((size_t)NB * CC * 2);
  half_t* W2cm    = (half_t*)alloc((size_t)CC * CC * 2);
  half_t* Wa1cm[LL]; half_t* Wvcm[LL]; half_t* Wocm[LL];
  half_t* Wf1cm[LL]; half_t* Wf2cm[LL];
  for (int l = 0; l < LL; ++l) {
    Wa1cm[l] = (half_t*)alloc((size_t)3 * CC * CC * 2);
    Wvcm[l]  = (half_t*)alloc((size_t)CC * CC * 2);
    Wocm[l]  = (half_t*)alloc((size_t)CC * CC * 2);
    Wf1cm[l] = (half_t*)alloc((size_t)CC * 512 * 2);
    Wf2cm[l] = (half_t*)alloc((size_t)512 * CC * 2);
  }

  auto cvt = [&](const float* w, half_t* o, int Kd, int Nd) {
    int n = Kd * Nd;
    k_w_cm<<<(n + 255) / 256, 256, 0, stream>>>(w, o, Kd, Nd);
  };
  cvt(W_rbf1, W1cm, NB, CC);
  cvt(W_rbf2, W2cm, CC, CC);
  for (int l = 0; l < LL; ++l) {
    cvt(Wa1 + (size_t)l * 3 * CC * CC, Wa1cm[l], 3 * CC, CC);
    cvt(Wv  + (size_t)l * CC * CC,     Wvcm[l],  CC, CC);
    cvt(Wo  + (size_t)l * CC * CC,     Wocm[l],  CC, CC);
    cvt(Wf1 + (size_t)l * CC * 512,    Wf1cm[l], CC, 512);
    cvt(Wf2 + (size_t)l * 512 * CC,    Wf2cm[l], 512, CC);
  }

  /* node init: x[:,0,:] = atom embedding, rest zero */
  k_node_init<<<(NN * CC + 255) / 256, 256, 0, stream>>>(atom_feats, atom_emb, xbuf);

  /* edge embedding: silu(rbf@W1+b1)@W2+b2 + bond embeddings -> e_f16 */
  k_gemm_rbf8<<<dim3(NE / 16, 1), 32, 0, stream>>>(edge_dist, W1cm, b_rbf1, edge_h);
  k_gemm_plain8<CC><<<dim3(NE / 16, 1), 32, 0, stream>>>(edge_h, W2cm, tmp, CC, 0);
  k_edge_final<<<(NE * CC + 255) / 256, 256, 0, stream>>>(tmp, b_rbf2, bond_emb,
                                                          bond_feats, e_f16);

  for (int l = 0; l < LL; ++l) {
    const float* Wa2l = Wa2 + (size_t)l * CC * HH;
    const float* Wgl  = Wg  + (size_t)l * CC * HH;

    /* ---- attention block ---- */
    k_rms_f16<<<NN, 256, 0, stream>>>(xbuf, xn_f16);
    k_gemm_attn18<<<dim3(NE / 16, 1), 32, 0, stream>>>(xn_f16, e_f16, src, dst,
                                                       Wa1cm[l], edge_h);
    k_seg_init<<<(NN * HH + 255) / 256, 256, 0, stream>>>(segord, segsum);
    k_zero_f32<<<(NKROWS * CC + 255) / 256, 256, 0, stream>>>(agg, NKROWS * CC);
    k_logits_segmax<<<(NE * HH + 255) / 256, 256, 0, stream>>>(edge_h, Wa2l, dst,
                                                               logits, segord);
    k_seg_expsum<<<(NE * HH + 255) / 256, 256, 0, stream>>>(logits, segord, dst,
                                                            scale, segsum);
    k_scale_final<<<(NE * HH + 255) / 256, 256, 0, stream>>>(e_f16, Wgl, dst,
                                                             segsum, scale);
    k_gemm_value8<<<dim3(RROWS / 16, 1), 32, 0, stream>>>(xn_f16, src, dst,
                                                          scale, Wvcm[l], agg);
    k_f32_to_f16<<<(NKROWS * CC + 255) / 256, 256, 0, stream>>>(agg, agg_f16,
                                                                NKROWS * CC);
    k_gemm_plain8<CC><<<dim3(NKROWS / 16, 1), 32, 0, stream>>>(agg_f16, Wocm[l],
                                                               xbuf, CC, 1);

    /* ---- equivariant FFN ---- */
    k_rms_f16<<<NN, 256, 0, stream>>>(xbuf, xn_f16);
    k_gemm_plain8<CC><<<dim3(NKROWS / 16, 4), 32, 0, stream>>>(xn_f16, Wf1cm[l],
                                                               tmp, 512, 0);
    k_ffn_act<<<(NN * 512 + 255) / 256, 256, 0, stream>>>(tmp, h_f16);
    k_gemm_plain8<NB><<<dim3(NKROWS / 16, 1), 32, 0, stream>>>(h_f16, Wf2cm[l],
                                                               xbuf, CC, 1);
  }

  /* final rms_norm_sh -> output */
  k_rms_out<<<NN, 256, 0, stream>>>(xbuf, (float*)d_out);
}
